// NAVARNet_67035849556364
// MI455X (gfx1250) — matrix operand order repaired and checked by reading, not compiled, not run on hardware
//
#include <hip/hip_runtime.h>

// NAVAR fused forward for MI455X (gfx1250, wave32, WMMA + async-to-LDS).
//
// out[b,v] = sum_{e,h} relu(x[b,e]*W1[e,h]+b1[e,h]) * W2[e,h,v] + sum_e b2[e,v]
//
// GEMM view: M=4096, N=112 (100 padded), K=E*H=64000, A generated on the fly
// in the WMMA bf16 A-register layout, B = pre-packed bf16 W2.
// - 128 WGs x 8 waves; each wave: 32 rows (2 A tiles), all 7 N tiles,
//   125 experts (K-split across waves, deterministic LDS reduction).
// - Per-expert operand block (W1,b1 f32 + 14 bf16 B tiles = 14848B) staged
//   into a per-wave double-buffered LDS ring with global_load_async_to_lds_b128
//   (ASYNCcnt), hiding L2 latency a full expert ahead.

#define NEXP   1000
#define HID    64
#define NV     100
#define BATCHN 4096
#define NT     7
#define KS     2
#define EBYTES 14848          // 512B (W1+b1 f32) + 14 * 1024B bf16 B tiles
#define ECHUNKS (EBYTES / 16) // 928 16-byte chunks
#define WAVES  8
#define EPW    (NEXP / WAVES) // 125 experts per wave

typedef __attribute__((ext_vector_type(16))) __bf16 v16bf;
typedef __attribute__((ext_vector_type(8)))  __bf16 v8bf;
typedef __attribute__((ext_vector_type(8)))  float  v8f;
typedef __attribute__((ext_vector_type(4)))  float  v4f;
typedef __attribute__((ext_vector_type(4)))  unsigned v4u;
typedef __attribute__((__vector_size__(4 * sizeof(int)))) int v4i;

static constexpr size_t BPACK_BYTES = (size_t)NEXP * EBYTES; // 14.848 MB
static constexpr size_t LDS_BYTES   = (size_t)WAVES * 2 * EBYTES; // 237568

#if __has_builtin(__builtin_amdgcn_global_load_async_to_lds_b128) && \
    __has_builtin(__builtin_amdgcn_s_wait_asynccnt)
#define HAVE_ASYNC_LDS 1
#endif

__device__ __forceinline__ void stage_expert(char* dst, const char* src, int lane) {
#if HAVE_ASYNC_LDS
#pragma unroll
  for (int c = 0; c < ECHUNKS / 32; ++c) {       // 29 async b128 per lane
    int off = (c * 32 + lane) * 16;
    __builtin_amdgcn_global_load_async_to_lds_b128(
        (v4i*)(src + off), (v4i*)(dst + off), 0, 0);
  }
#else
  for (int c = lane; c < ECHUNKS; c += 32) {
    v4u d = *(const v4u*)(src + c * 16);
    *(v4u*)(dst + c * 16) = d;
  }
#endif
}

__device__ __forceinline__ void wait_async0() {
#if HAVE_ASYNC_LDS
  __builtin_amdgcn_s_wait_asynccnt(0);
  __asm__ volatile("" ::: "memory");
#endif
}

// ---------------------------------------------------------------------------
// Pack kernels: per-expert block at ws + e*EBYTES:
//   [0,256)   W1[e] f32   [256,512) b1[e] f32
//   [512 + (t*2+s)*1024, ...): 32x16 bf16 B tile, sub0 = halves0..7 lane*16,
//                              sub1(+512) = halves8..15 lane*16.
// Per-lane B layout: col n = L%16; K = (L<16?0:16) + half.
// ---------------------------------------------------------------------------
__global__ void pack_w2(const float* __restrict__ W2, char* __restrict__ ws) {
  int tid = blockIdx.x * blockDim.x + threadIdx.x;
  if (tid >= NEXP * NT * KS * 32) return;
  int lane = tid & 31;
  int blk  = tid >> 5;                    // (e*NT + t)*KS + s
  int s = blk % KS;
  int e = blk / (KS * NT);
  int t = (blk / KS) % NT;
  int n    = t * 16 + (lane & 15);
  int kofs = (lane < 16) ? 0 : 16;
  __bf16* base = (__bf16*)(ws + (size_t)e * EBYTES + 512 + (size_t)(t * KS + s) * 1024);

  v8bf lo, hi;
#pragma unroll
  for (int j = 0; j < 8; ++j) {
    int hh = s * 32 + kofs + j;
    lo[j] = (__bf16)((n < NV) ? W2[((size_t)e * HID + hh) * NV + n] : 0.0f);
  }
#pragma unroll
  for (int j = 0; j < 8; ++j) {
    int hh = s * 32 + kofs + 8 + j;
    hi[j] = (__bf16)((n < NV) ? W2[((size_t)e * HID + hh) * NV + n] : 0.0f);
  }
  *(v8bf*)(base + lane * 8)       = lo;
  *(v8bf*)(base + 256 + lane * 8) = hi;
}

__global__ void copy_w1b1(const float* __restrict__ W1, const float* __restrict__ b1,
                          char* __restrict__ ws) {
  int tid = blockIdx.x * blockDim.x + threadIdx.x;
  if (tid >= NEXP * 128) return;
  int e = tid >> 7, i = tid & 127;
  float v = (i < HID) ? W1[(size_t)e * HID + i] : b1[(size_t)e * HID + i - HID];
  ((float*)(ws + (size_t)e * EBYTES))[i] = v;
}

__global__ void b2_reduce(const float* __restrict__ b2, float* __restrict__ b2sum) {
  int v = blockIdx.x * blockDim.x + threadIdx.x;
  if (v >= NV) return;
  float s = 0.0f;
  for (int e = 0; e < NEXP; ++e) s += b2[(size_t)e * NV + v];
  b2sum[v] = s;
}

// ---------------------------------------------------------------------------
// Fused kernel. 256 threads = 8 waves. Wave w: rows [blk*32, blk*32+32),
// experts [w*125, (w+1)*125). Epilogue: deterministic 8-way LDS reduction.
// ---------------------------------------------------------------------------
extern "C" __global__ void __launch_bounds__(256)
navar_fused(const float* __restrict__ x, const char* __restrict__ bp,
            const float* __restrict__ b2sum, float* __restrict__ out) {
  extern __shared__ char smem[];
  const int lane  = threadIdx.x & 31;
  const int wave  = threadIdx.x >> 5;
  const int mbase = blockIdx.x * 32;
  const int lhi   = lane >> 4;
  const int klo   = lhi * 16;
  const int row0  = mbase + (lane & 15);
  const int row1  = row0 + 16;
  char* wbuf = smem + (size_t)wave * (2 * EBYTES);
  const int e0 = wave * EPW, e1 = e0 + EPW;

  v8f acc[2][NT] = {};

  stage_expert(wbuf, bp + (size_t)e0 * EBYTES, lane);
  float xv0 = x[(size_t)row0 * NEXP + e0];
  float xv1 = x[(size_t)row1 * NEXP + e0];

  for (int e = e0; e < e1; ++e) {
    const int p = (e - e0) & 1;
    char* cur = wbuf + p * EBYTES;
    wait_async0();                              // current expert staged
    if (e + 1 < e1)                             // prefetch next expert
      stage_expert(wbuf + (p ^ 1) * EBYTES, bp + (size_t)(e + 1) * EBYTES, lane);
    float xn0 = (e + 1 < e1) ? x[(size_t)row0 * NEXP + e + 1] : 0.0f;
    float xn1 = (e + 1 < e1) ? x[(size_t)row1 * NEXP + e + 1] : 0.0f;

    const float* w1l = (const float*)cur;
    const float* b1l = w1l + HID;

#pragma unroll
    for (int s = 0; s < KS; ++s) {
      const int h0 = s * 32 + klo;
      const int h1 = h0 + 8;
      v4f wa = *(const v4f*)(w1l + h0);
      v4f wb = *(const v4f*)(w1l + h0 + 4);
      v4f wc = *(const v4f*)(w1l + h1);
      v4f wd = *(const v4f*)(w1l + h1 + 4);
      v4f ba = *(const v4f*)(b1l + h0);
      v4f bb = *(const v4f*)(b1l + h0 + 4);
      v4f bc = *(const v4f*)(b1l + h1);
      v4f bd = *(const v4f*)(b1l + h1 + 4);

      v16bf a0, a1;
#pragma unroll
      for (int j = 0; j < 4; ++j) {
        a0[j]      = (__bf16)fmaxf(fmaf(xv0, wa[j], ba[j]), 0.0f);
        a0[4 + j]  = (__bf16)fmaxf(fmaf(xv0, wb[j], bb[j]), 0.0f);
        a0[8 + j]  = (__bf16)fmaxf(fmaf(xv0, wc[j], bc[j]), 0.0f);
        a0[12 + j] = (__bf16)fmaxf(fmaf(xv0, wd[j], bd[j]), 0.0f);
        a1[j]      = (__bf16)fmaxf(fmaf(xv1, wa[j], ba[j]), 0.0f);
        a1[4 + j]  = (__bf16)fmaxf(fmaf(xv1, wb[j], bb[j]), 0.0f);
        a1[8 + j]  = (__bf16)fmaxf(fmaf(xv1, wc[j], bc[j]), 0.0f);
        a1[12 + j] = (__bf16)fmaxf(fmaf(xv1, wd[j], bd[j]), 0.0f);
      }

#pragma unroll
      for (int t = 0; t < NT; ++t) {
        const char* blkb = cur + 512 + (size_t)(t * KS + s) * 1024;
        v8bf blo = *(const v8bf*)(blkb + lane * 16);
        v8bf bhi = *(const v8bf*)(blkb + 512 + lane * 16);
        v16bf bmat = __builtin_shufflevector(blo, bhi,
                       0, 1, 2, 3, 4, 5, 6, 7, 8, 9, 10, 11, 12, 13, 14, 15);
        acc[0][t] = __builtin_amdgcn_wmma_f32_16x16x32_bf16(
            false, a0, false, bmat, (short)0, acc[0][t], false, false);
        acc[1][t] = __builtin_amdgcn_wmma_f32_16x16x32_bf16(
            false, a1, false, bmat, (short)0, acc[1][t], false, false);
      }
    }
    xv0 = xn0; xv1 = xn1;
  }

  // ----- deterministic 8-wave reduction through (reused) LDS -----
  wait_async0();
  __syncthreads();
  float* red = (float*)smem;   // wave w region: [w*3584, (w+1)*3584) floats
#pragma unroll
  for (int m2 = 0; m2 < 2; ++m2)
#pragma unroll
    for (int t = 0; t < NT; ++t)
#pragma unroll
      for (int r = 0; r < 8; ++r)
        red[wave * 3584 + ((m2 * NT + t) * 8 + r) * 32 + lane] = acc[m2][t][r];
  __syncthreads();

  const int tid = threadIdx.x;
#pragma unroll
  for (int j = 0; j < 2 * NT; ++j) {
    float s = 0.0f;
#pragma unroll
    for (int w = 0; w < WAVES; ++w) s += red[w * 3584 + j * 256 + tid];
    int m2 = j / NT, t = j % NT;
    int r = tid >> 5, ln = tid & 31;
    int col = t * 16 + (ln & 15);
    int row = mbase + m2 * 16 + (ln >> 4) * 8 + r;
    if (col < NV) out[(size_t)row * NV + col] = s + b2sum[col];
  }
}

// ---------------------------------------------------------------------------
extern "C" void kernel_launch(void* const* d_in, const int* in_sizes, int n_in,
                              void* d_out, int out_size, void* d_ws, size_t ws_size,
                              hipStream_t stream) {
  (void)in_sizes; (void)n_in; (void)out_size; (void)ws_size;
  const float* x  = (const float*)d_in[0];
  const float* W1 = (const float*)d_in[1];
  const float* b1 = (const float*)d_in[2];
  const float* W2 = (const float*)d_in[3];
  const float* b2 = (const float*)d_in[4];
  float* out = (float*)d_out;

  char*  ws    = (char*)d_ws;
  float* b2sum = (float*)(ws + BPACK_BYTES);

  (void)hipFuncSetAttribute((const void*)navar_fused,
                            hipFuncAttributeMaxDynamicSharedMemorySize,
                            (int)LDS_BYTES);

  pack_w2<<<(NEXP * NT * KS * 32 + 255) / 256, 256, 0, stream>>>(W2, ws);
  copy_w1b1<<<(NEXP * 128 + 255) / 256, 256, 0, stream>>>(W1, b1, ws);
  b2_reduce<<<1, 128, 0, stream>>>(b2, b2sum);
  navar_fused<<<BATCHN / 32, 256, LDS_BYTES, stream>>>(x, ws, b2sum, out);
}
